// Net_28956669510106
// MI455X (gfx1250) — matrix operand order repaired and checked by reading, not compiled, not run on hardware
//
#include <hip/hip_runtime.h>
#include <hip/hip_bf16.h>

// ---------------------------------------------------------------------------
// Sizes (fixed by the reference):
//   B=16, NF=3, C=128, H=W=56, GRID=28, K=128.  vb = b*NF + v  (48 values)
//   Output: [B, NF, K, 28, 28] f32 = (vb*128 + k)*784 + hw
// Workspace layout (bf16):
//   packedA: [vb=48][mt=8][s=4][lane=32][16]     (f2 gathered, A-fragments)
//   packedB: [vb=48][nt=49][s=4][lane=32][16]    (f1 downsampled, B-fragments)
// ---------------------------------------------------------------------------

typedef __attribute__((ext_vector_type(16))) __bf16 v16bf;
typedef __attribute__((ext_vector_type(8)))  float  v8f;
typedef __attribute__((ext_vector_type(4)))  int    v4i;

#define NVB   48
#define NC    128
#define HW_IN 3136   // 56*56
#define NQ    784    // 28*28
#define NK    128
#define NTILE 49     // 784/16
#define SCALE_56_28 (55.0f / 27.0f)

// ---- CDNA5 async global->LDS copy (ASYNCcnt path), with safe fallback ------
#if defined(__has_builtin)
#if __has_builtin(__builtin_amdgcn_global_load_async_to_lds_b128)
#define USE_ASYNC_LDS 1
#endif
#if __has_builtin(__builtin_amdgcn_s_wait_asynccnt)
#define HAVE_WAIT_ASYNC 1
#endif
#endif

#define GLOBAL_AS __attribute__((address_space(1)))
#define LDS_AS    __attribute__((address_space(3)))

__device__ __forceinline__ void copy16(const void* g, void* l) {
#if defined(USE_ASYNC_LDS)
  __builtin_amdgcn_global_load_async_to_lds_b128(
      (GLOBAL_AS v4i*)g, (LDS_AS v4i*)l, /*offset=*/0, /*cpol=*/0);
#else
  *(ulonglong2*)l = *(const ulonglong2*)g;
#endif
}

#if defined(USE_ASYNC_LDS)
#if defined(HAVE_WAIT_ASYNC)
#define ASYNC_WAIT(n) __builtin_amdgcn_s_wait_asynccnt(n)
#else
#define ASYNC_WAIT(n) asm volatile("s_wait_asynccnt %0" :: "i"(n) : "memory")
#endif
#else
#define ASYNC_WAIT(n) ((void)0)
#endif

// ---------------------------------------------------------------------------
// Kernel 1: bilinear downsample (align_corners) + bf16 convert + fragment pack
// grid = 48 * 32 blocks (vb, channel-group-of-4), 256 threads
// ---------------------------------------------------------------------------
__device__ __forceinline__ void stage_plane(const float* __restrict__ g,
                                            float* plane, int tid) {
  // 3136 floats = 784 x 16B; 784 = 3*256 + 16
  #pragma unroll
  for (int it = 0; it < 3; ++it) {
    const int i = tid + it * 256;
    copy16(g + i * 4, plane + i * 4);
  }
  if (tid < 16) {
    const int i = 768 + tid;
    copy16(g + i * 4, plane + i * 4);
  }
}

__device__ __forceinline__ float bilerp28(const float* plane, int oi, int oj) {
  float sy = oi * SCALE_56_28;
  int   i0 = (int)sy; if (i0 > 54) i0 = 54;
  float wy = sy - (float)i0;
  float sx = oj * SCALE_56_28;
  int   j0 = (int)sx; if (j0 > 54) j0 = 54;
  float wx = sx - (float)j0;
  const float* r0 = &plane[i0 * 56 + j0];
  return (1.f - wy) * ((1.f - wx) * r0[0]  + wx * r0[1])
       +        wy  * ((1.f - wx) * r0[56] + wx * r0[57]);
}

__global__ __launch_bounds__(256) void pack_kernel(
    const float* __restrict__ fi, const float* __restrict__ fj,
    const int* __restrict__ knn,          // [NF, K, 2] (int32 per harness)
    __bf16* __restrict__ packedA, __bf16* __restrict__ packedB)
{
  __shared__ __align__(16) float plane[HW_IN];
  __shared__ int knnS[NK * 2];

  const int vb  = blockIdx.x >> 5;
  const int cg  = blockIdx.x & 31;      // group of 4 channels
  const int v   = vb % 3;
  const int tid = threadIdx.x;

  knnS[tid] = knn[v * NK * 2 + tid];    // 256 threads load exactly K*2 ints

  #pragma unroll 1
  for (int ch = 0; ch < 4; ++ch) {
    const int c = cg * 4 + ch;
    const int s = c >> 5;               // k-step (32 channels each)
    const int L = c & 31;               // k offset within the k-step

    const float* p1 = fi + ((size_t)vb * NC + c) * HW_IN;
    const float* p2 = fj + ((size_t)vb * NC + c) * HW_IN;

    // pull next channel's planes toward L2 while we work on this one
    if (ch < 3) {
      __builtin_prefetch(p1 + HW_IN + tid * 12, 0, 0);
      __builtin_prefetch(p2 + HW_IN + tid * 12, 0, 0);
    }

    // ---------------- f1: full 28x28 downsample -> B fragments --------------
    __syncthreads();                    // prior plane consumers done
    stage_plane(p1, plane, tid);
    ASYNC_WAIT(0);
    __syncthreads();

    #pragma unroll
    for (int it = 0; it < 4; ++it) {    // 784 = 3*256 + 16
      const int o = tid + it * 256;
      if (it < 3 || tid < 16) {
        const int oi = o / 28, oj = o - oi * 28;
        const float val = bilerp28(plane, oi, oj);
        const int nt = o >> 4, e = o & 15;
        packedB[(((size_t)vb * NTILE + nt) * 4 + s) * 512 + (e + (((L >> 3) & 1) << 4)) * 16 + (L & 7) + ((L >> 4) << 3)] = (__bf16)val;
      }
    }

    // ---------------- f2: gather at K knn grid points -> A fragments --------
    __syncthreads();
    stage_plane(p2, plane, tid);
    ASYNC_WAIT(0);
    __syncthreads();

    if (tid < NK) {
      const int k  = tid;
      const int gx = knnS[k * 2 + 0];   // q = knn[:,1]*28 + knn[:,0]
      const int gy = knnS[k * 2 + 1];
      const float val = bilerp28(plane, gy, gx);

      // A fragment layout (16-bit A 16x32):
      //   lane<16 : row M=lane,    K-offsets {0..7, 16..23}
      //   lane>=16: row M=lane-16, K-offsets {8..15, 24..31}
      const int mt = k >> 4, r = k & 15, co = c & 31;
      int Ln, e;
      if      (co <  8) { Ln = r;      e = co;      }
      else if (co < 16) { Ln = r + 16; e = co - 8;  }
      else if (co < 24) { Ln = r;      e = co - 8;  }
      else              { Ln = r + 16; e = co - 16; }
      packedA[(((size_t)vb * 8 + mt) * 4 + s) * 512 + Ln * 16 + e] = (__bf16)val;
    }
  }
}

// ---------------------------------------------------------------------------
// Kernel 2: 128x784 GEMM (C=128 reduction) via bf16 WMMA + relu/exp-normalize
// grid = 48 * 8 blocks (vb, m-tile), 224 threads = 7 waves
// Double-buffered B chunks staged with async global->LDS copies.
// ---------------------------------------------------------------------------
#define WAVES 7
#define TPB   (WAVES * 32)

__global__ __launch_bounds__(TPB) void gemm_softmax_kernel(
    const __bf16* __restrict__ packedA, const __bf16* __restrict__ packedB,
    float* __restrict__ out)
{
  __shared__ __align__(16) __bf16 Alds[4 * 512];              // 4 KB
  __shared__ __align__(16) __bf16 Blds[2][WAVES * 4 * 512];   // 2 x 28 KB
  __shared__ float red[WAVES * 16];
  __shared__ float denomLds[16];

  const int vb   = blockIdx.x >> 3;
  const int mt   = blockIdx.x & 7;
  const int tid  = threadIdx.x;
  const int wave = tid >> 5;
  const int lane = tid & 31;

  const __bf16* baseB = packedB + (size_t)vb * NTILE * 2048;

  // ---- stage A fragments (2048 bf16 = 256 x 16B; 256 = 224 + 32) ----
  {
    const __bf16* srcA = packedA + ((size_t)(vb * 8 + mt)) * 2048;
    copy16(srcA + tid * 8, Alds + tid * 8);
    if (tid < 32) copy16(srcA + (tid + TPB) * 8, Alds + (tid + TPB) * 8);
  }
  // ---- stage B chunk 0 (7 tiles * 2048 bf16 = 1792 x 16B = 8 per thread) ----
  #pragma unroll
  for (int it = 0; it < 8; ++it) {
    const int i = tid + it * TPB;
    copy16(baseB + /*chunk0*/ i * 8, &Blds[0][i * 8]);
  }
  ASYNC_WAIT(0);
  __syncthreads();

  v16bf aFrag[4];
  #pragma unroll
  for (int s = 0; s < 4; ++s)
    aFrag[s] = *(const v16bf*)&Alds[s * 512 + lane * 16];

  v8f zero = {};
  v8f acc[7];
  #pragma unroll
  for (int c = 0; c < 7; ++c) acc[c] = zero;

  // ---- 7 chunks of 7 n-tiles; wave w owns tile (chunk*7 + w) ----
  // Fully unrolled so acc[] stays in VGPRs (runtime indexing would spill).
  #pragma unroll
  for (int chunk = 0; chunk < 7; ++chunk) {
    const int cur = chunk & 1;
    if (chunk < 6) {
      // prefetch next chunk into the other buffer (its last readers finished
      // before the barrier that ended the previous iteration)
      const __bf16* srcB = baseB + (size_t)(chunk + 1) * 7 * 2048;
      #pragma unroll
      for (int it = 0; it < 8; ++it) {
        const int i = tid + it * TPB;
        copy16(srcB + i * 8, &Blds[cur ^ 1][i * 8]);
      }
      ASYNC_WAIT(8);   // oldest 8 (current chunk) complete; next 8 in flight
    } else {
      ASYNC_WAIT(0);
    }
    __syncthreads();   // current chunk visible to all waves

    const __bf16* bbase = &Blds[cur][wave * 2048];
    v8f a = acc[chunk];
    #pragma unroll
    for (int s = 0; s < 4; ++s) {
      v16bf bFrag = *(const v16bf*)&bbase[s * 512 + lane * 16];
      a = __builtin_amdgcn_wmma_f32_16x16x32_bf16(
              /*neg_a=*/false, aFrag[s], /*neg_b=*/false, bFrag,
              /*c_mod=*/(short)0, a, /*reuse_a=*/false, /*reuse_b=*/false);
    }
    acc[chunk] = a;
    __syncthreads();   // all waves done reading Blds[cur] before overwrite
  }

  // ---- epilogue: relu, per-row exp-sum, normalize, store ----
  // C/D layout: VGPR r of lane L = element (row = r + 8*(L>=16), col = L%16)
  float esum[8];
  #pragma unroll
  for (int r = 0; r < 8; ++r) esum[r] = 0.f;

  #pragma unroll
  for (int c = 0; c < 7; ++c) {
    v8f vv = acc[c];
    #pragma unroll
    for (int r = 0; r < 8; ++r) {
      float x = vv[r] > 0.f ? vv[r] : 0.f;
      vv[r] = x;
      esum[r] += __expf(x);
    }
    acc[c] = vv;
  }

  // reduce across the 16 columns held in each 16-lane half (xor 1..8 stays in-half)
  #pragma unroll
  for (int off = 1; off < 16; off <<= 1)
    #pragma unroll
    for (int r = 0; r < 8; ++r)
      esum[r] += __shfl_xor(esum[r], off, 32);

  if (lane == 0) {
    #pragma unroll
    for (int r = 0; r < 8; ++r) red[wave * 16 + r] = esum[r];
  }
  if (lane == 16) {
    #pragma unroll
    for (int r = 0; r < 8; ++r) red[wave * 16 + 8 + r] = esum[r];
  }
  __syncthreads();

  if (tid < 16) {
    float d = 0.f;
    #pragma unroll
    for (int w = 0; w < WAVES; ++w) d += red[w * 16 + tid];
    denomLds[tid] = d;
  }
  __syncthreads();

  const int hi  = lane >> 4;      // 0: rows 0..7, 1: rows 8..15
  const int col = lane & 15;
  float dscale[8];
  #pragma unroll
  for (int r = 0; r < 8; ++r) dscale[r] = 10.f / denomLds[hi * 8 + r];

  #pragma unroll
  for (int c = 0; c < 7; ++c) {
    const int nt = c * 7 + wave;
    const int hw = nt * 16 + col;
    v8f vv = acc[c];
    #pragma unroll
    for (int r = 0; r < 8; ++r) {
      const int krow = mt * 16 + hi * 8 + r;
      out[((size_t)vb * NK + krow) * NQ + hw] = vv[r] * dscale[r];
    }
  }
}

// ---------------------------------------------------------------------------
extern "C" void kernel_launch(void* const* d_in, const int* in_sizes, int n_in,
                              void* d_out, int out_size, void* d_ws, size_t ws_size,
                              hipStream_t stream) {
  const float* fi  = (const float*)d_in[0];   // feature_i [16,3,128,56,56]
  const float* fj  = (const float*)d_in[1];   // feature_j [16,3,128,56,56]
  const int*   knn = (const int*)d_in[4];     // knn_inds  [3,128,2]
  float* out = (float*)d_out;                 // [16,3,128,28,28]

  __bf16* packedA = (__bf16*)d_ws;                       // 1.5 MB
  __bf16* packedB = packedA + (size_t)NVB * 8 * 4 * 512; // 9.2 MB

  pack_kernel<<<NVB * 32, 256, 0, stream>>>(fi, fj, knn, packedA, packedB);
  gemm_softmax_kernel<<<NVB * 8, TPB, 0, stream>>>(packedA, packedB, out);
}
